// Model_2147483648356
// MI455X (gfx1250) — matrix operand order repaired and checked by reading, not compiled, not run on hardware
//
#include <hip/hip_runtime.h>
#include <math.h>

// ---------------- types for CDNA5 WMMA ----------------
typedef __bf16 bf16_t;
typedef __attribute__((ext_vector_type(16))) bf16_t bf16x16;
typedef __attribute__((ext_vector_type(8)))  bf16_t bf16x8;
typedef __attribute__((ext_vector_type(8)))  float  f32x8;

#define DV   512     // model dim D
#define NV   7       // variants V
#define NB   8       // batch B
#define NT   512     // time T
#define NH   4       // heads
#define DH   128     // head dim

static __device__ __forceinline__ bf16_t f2bf(float f) {
    unsigned int u = __builtin_bit_cast(unsigned int, f);
    u += 0x7FFFu + ((u >> 16) & 1u);     // round-to-nearest-even
    unsigned short h = (unsigned short)(u >> 16);
    return __builtin_bit_cast(bf16_t, h);
}
static __device__ __forceinline__ bf16x8 pack8(float4 a, float4 b) {
    bf16x8 r;
    r[0]=f2bf(a.x); r[1]=f2bf(a.y); r[2]=f2bf(a.z); r[3]=f2bf(a.w);
    r[4]=f2bf(b.x); r[5]=f2bf(b.y); r[6]=f2bf(b.z); r[7]=f2bf(b.w);
    return r;
}

// f32 -> bf16 bulk conversion (8 elems/thread); launch with elems/2048 blocks
__global__ __launch_bounds__(256) void cvt_f32_bf16(
    const float* __restrict__ in, bf16_t* __restrict__ out)
{
    int i = (blockIdx.x * 256 + threadIdx.x) * 8;
    float4 a = *(const float4*)(in + i);
    float4 b = *(const float4*)(in + i + 4);
    *(bf16x8*)(out + i) = pack8(a, b);
}

// =====================================================================
// NT GEMM, bf16 operands, f32 accumulate:  C[M,N] = A[M,K] * W[N,K]^T + bias
// A, W pre-converted to bf16. M%128==0, N%64==0, K%32==0 (always true here).
// 256 threads = 8 waves; block tile 128x64; K step 32; wave = 2x2 wmma tiles.
// Tiles staged global->LDS with GLOBAL_LOAD_ASYNC_TO_LDS_B128, double
// buffered and pipelined on ASYNCcnt (3 async b128 per thread per K-step).
// =====================================================================
template<bool RELU>
__global__ __launch_bounds__(256) void gemm_nt_bias(
    const bf16_t* __restrict__ A, int lda,
    const bf16_t* __restrict__ W, int ldw,
    const float* __restrict__ bias,
    float* __restrict__ C, int ldc,
    int K)
{
    constexpr int LS = 40;                 // padded LDS row stride (bf16): 80B, 16B-aligned
    __shared__ bf16_t sA[2][128 * LS];
    __shared__ bf16_t sB[2][64 * LS];

    const int tid  = threadIdx.x;
    const int lane = tid & 31;
    const int wave = tid >> 5;
    const int mBase = (wave >> 1) * 32;
    const int nBase = (wave & 1) * 32;
    const int blockM = blockIdx.y * 128;
    const int blockN = blockIdx.x * 64;
    const int l15 = lane & 15;
    const int lhi = lane >> 4;

    // seed accumulators with bias (col depends only on j,lane)
    f32x8 acc[2][2];
    #pragma unroll
    for (int j = 0; j < 2; ++j) {
        float bj = bias[blockN + nBase + j * 16 + l15];
        #pragma unroll
        for (int i = 0; i < 2; ++i)
            #pragma unroll
            for (int g = 0; g < 8; ++g) acc[i][j][g] = bj;
    }

    // async staging: thread owns two A 16B chunks (rows cr, cr+64) + one W chunk
    const int cr = tid >> 2;               // 0..63
    const int cs = (tid & 3) * 8;          // k-seg (bf16 elems): 0,8,16,24
    const unsigned long long gA0 =
        (unsigned long long)(A + (size_t)(blockM + cr) * lda + cs);
    const unsigned long long gA1 =
        (unsigned long long)(A + (size_t)(blockM + cr + 64) * lda + cs);
    const unsigned long long gB =
        (unsigned long long)(W + (size_t)(blockN + cr) * ldw + cs);
    unsigned lA0[2], lA1[2], lB[2];
    #pragma unroll
    for (int b = 0; b < 2; ++b) {
        lA0[b] = (unsigned)(size_t)&sA[b][cr * LS + cs];
        lA1[b] = (unsigned)(size_t)&sA[b][(cr + 64) * LS + cs];
        lB[b]  = (unsigned)(size_t)&sB[b][cr * LS + cs];
    }

    auto stage = [&](int k0, int b) {
        unsigned long long off = (unsigned long long)k0 * 2;   // bytes
        asm volatile("global_load_async_to_lds_b128 %0, %1, off"
                     :: "v"(lA0[b]), "v"(gA0 + off) : "memory");
        asm volatile("global_load_async_to_lds_b128 %0, %1, off"
                     :: "v"(lA1[b]), "v"(gA1 + off) : "memory");
        asm volatile("global_load_async_to_lds_b128 %0, %1, off"
                     :: "v"(lB[b]),  "v"(gB  + off) : "memory");
    };

    stage(0, 0);
    if (K > 32) stage(32, 1);

    int buf = 0;
    for (int k0 = 0; k0 < K; k0 += 32) {
        // oldest 3 async ops (current buf) must land; newer buffer stays in flight
        if (k0 + 32 < K) asm volatile("s_wait_asynccnt 3" ::: "memory");
        else             asm volatile("s_wait_asynccnt 0" ::: "memory");
        __syncthreads();

        // A fragment: lane = M row (l15); lanes 0-15 K{0-7,16-23}, 16-31 K{8-15,24-31}
        bf16x16 af[2], bfr[2];
        #pragma unroll
        for (int i = 0; i < 2; ++i) {
            int m = mBase + i * 16 + l15;
            bf16x8 lo = *(const bf16x8*)&sA[buf][m * LS + lhi * 8];
            bf16x8 hi = *(const bf16x8*)&sA[buf][m * LS + 16 + lhi * 8];
            #pragma unroll
            for (int e = 0; e < 8; ++e) { af[i][e] = lo[e]; af[i][e + 8] = hi[e]; }
        }
        // B fragment: lane = N col (l15); lanes 0-15 K{0-7,16-23}, 16-31 K{8-15,24-31}
        #pragma unroll
        for (int j = 0; j < 2; ++j) {
            int n = nBase + j * 16 + l15;
            bf16x8 lo = *(const bf16x8*)&sB[buf][n * LS + lhi * 8];
            bf16x8 hi = *(const bf16x8*)&sB[buf][n * LS + 16 + lhi * 8];
            #pragma unroll
            for (int e = 0; e < 8; ++e) { bfr[j][e] = lo[e]; bfr[j][e + 8] = hi[e]; }
        }
        #pragma unroll
        for (int i = 0; i < 2; ++i)
            #pragma unroll
            for (int j = 0; j < 2; ++j)
                acc[i][j] = __builtin_amdgcn_wmma_f32_16x16x32_bf16(
                    false, af[i], false, bfr[j], (short)0, acc[i][j], false, false);

        if (k0 + 64 < K) {
            __syncthreads();          // all waves done reading buf before refill
            stage(k0 + 64, buf);
        }
        buf ^= 1;
    }

    // C/D layout: VGPR g -> lanes 0-15 M=g, lanes 16-31 M=g+8; N = lane&15
    #pragma unroll
    for (int i = 0; i < 2; ++i)
        #pragma unroll
        for (int j = 0; j < 2; ++j) {
            int col = blockN + nBase + j * 16 + l15;
            int row0 = blockM + mBase + i * 16 + lhi * 8;
            float* cp = C + (size_t)row0 * ldc + col;
            #pragma unroll
            for (int g = 0; g < 8; ++g) {
                float v = acc[i][j][g];
                if (RELU) v = fmaxf(v, 0.0f);
                cp[(size_t)g * ldc] = v;
            }
        }
}

// =====================================================================
// Per-variant embed: toks[bt,v,d] = x[bt,v]*ve_w[v,d] + ve_b[v,d]
// =====================================================================
__global__ __launch_bounds__(256) void embed_kernel(
    const float* __restrict__ x, const float* __restrict__ vew,
    const float* __restrict__ veb, float* __restrict__ out)
{
    int idx = blockIdx.x * 256 + threadIdx.x;
    int d = idx & 511;
    int r = idx >> 9;            // bt*7 + v
    int v = r % NV;
    out[idx] = x[r] * vew[v * DV + d] + veb[v * DV + d];
}

// =====================================================================
// Variant attention: L=7 sequences, one block per (bt, head)
// =====================================================================
__global__ __launch_bounds__(128) void vattn_kernel(
    const float* __restrict__ qkv, float* __restrict__ out, float scale)
{
    __shared__ float sQ[NV][DH], sK[NV][DH], sV[NV][DH];
    __shared__ float sP[NV][8];
    const int bt = blockIdx.x, h = blockIdx.y, tid = threadIdx.x;
    const size_t base = (size_t)bt * NV * 1536 + h * DH;
    for (int idx = tid; idx < NV * DH; idx += 128) {
        int i = idx >> 7, d = idx & 127;
        sQ[i][d] = qkv[base + (size_t)i * 1536 + d];
        sK[i][d] = qkv[base + (size_t)i * 1536 + 512 + d];
        sV[i][d] = qkv[base + (size_t)i * 1536 + 1024 + d];
    }
    __syncthreads();
    if (tid < 49) {
        int i = tid / 7, j = tid % 7;
        float s = 0.f;
        #pragma unroll 8
        for (int d = 0; d < DH; ++d) s += sQ[i][d] * sK[j][d];
        sP[i][j] = s * scale;
    }
    __syncthreads();
    if (tid < 7) {
        float m = -1e30f;
        for (int j = 0; j < 7; ++j) m = fmaxf(m, sP[tid][j]);
        float p[7], l = 0.f;
        for (int j = 0; j < 7; ++j) { p[j] = __expf(sP[tid][j] - m); l += p[j]; }
        float inv = 1.0f / l;
        for (int j = 0; j < 7; ++j) sP[tid][j] = p[j] * inv;
    }
    __syncthreads();
    const size_t ob = (size_t)bt * NV * DV + h * DH;
    for (int idx = tid; idx < NV * DH; idx += 128) {
        int i = idx >> 7, d = idx & 127;
        float a = 0.f;
        #pragma unroll
        for (int j = 0; j < 7; ++j) a += sP[i][j] * sV[j][d];
        out[ob + (size_t)i * DV + d] = a;
    }
}

// =====================================================================
// Flash attention (f32, online softmax). One wave per query row.
// =====================================================================
__global__ __launch_bounds__(256) void flash_kernel(
    const float* __restrict__ Q, int ldq, long long strQ,
    const float* __restrict__ K, int ldk, long long strK,
    const float* __restrict__ Vv, int ldv, long long strV,
    float* __restrict__ O, int ldo, long long strO,
    int Lk, float scale)
{
    const int lane = threadIdx.x & 31;
    const int qi = blockIdx.x * 8 + (threadIdx.x >> 5);
    const int h = blockIdx.y, z = blockIdx.z;
    const float* qp = Q + (size_t)z * strQ + (size_t)qi * ldq + h * DH;
    const float* kp = K + (size_t)z * strK + h * DH;
    const float* vp = Vv + (size_t)z * strV + h * DH;
    float4 q4 = *(const float4*)(qp + lane * 4);
    float4 o4 = {0.f, 0.f, 0.f, 0.f};
    float m = -1e30f, l = 0.f;
    for (int j = 0; j < Lk; ++j) {
        float4 k4 = *(const float4*)(kp + (size_t)j * ldk + lane * 4);
        float s = q4.x * k4.x + q4.y * k4.y + q4.z * k4.z + q4.w * k4.w;
        #pragma unroll
        for (int off = 16; off > 0; off >>= 1) s += __shfl_xor(s, off, 32);
        s *= scale;
        float mn = fmaxf(m, s);
        float corr = __expf(m - mn);
        float p = __expf(s - mn);
        float4 v4 = *(const float4*)(vp + (size_t)j * ldv + lane * 4);
        o4.x = o4.x * corr + p * v4.x;
        o4.y = o4.y * corr + p * v4.y;
        o4.z = o4.z * corr + p * v4.z;
        o4.w = o4.w * corr + p * v4.w;
        l = l * corr + p;
        m = mn;
    }
    float inv = 1.0f / l;
    float4 r = {o4.x * inv, o4.y * inv, o4.z * inv, o4.w * inv};
    *(float4*)(O + (size_t)z * strO + (size_t)qi * ldo + h * DH + lane * 4) = r;
}

// =====================================================================
// Fused residual add + LayerNorm over D=512. One wave per row.
// =====================================================================
__global__ __launch_bounds__(256) void add_ln_kernel(
    const float* __restrict__ x, const float* __restrict__ y,
    const float* __restrict__ g, const float* __restrict__ b,
    float* __restrict__ out)
{
    const int lane = threadIdx.x & 31;
    const int row = blockIdx.x * 8 + (threadIdx.x >> 5);
    const size_t base = (size_t)row * DV;
    float v[16], s = 0.f;
    #pragma unroll
    for (int k = 0; k < 16; ++k) {
        int c = k * 32 + lane;
        v[k] = x[base + c] + y[base + c];
        s += v[k];
    }
    #pragma unroll
    for (int off = 16; off > 0; off >>= 1) s += __shfl_xor(s, off, 32);
    float mean = s * (1.0f / 512.0f);
    float vs = 0.f;
    #pragma unroll
    for (int k = 0; k < 16; ++k) { float d = v[k] - mean; vs += d * d; }
    #pragma unroll
    for (int off = 16; off > 0; off >>= 1) vs += __shfl_xor(vs, off, 32);
    float rstd = rsqrtf(vs * (1.0f / 512.0f) + 1e-5f);
    #pragma unroll
    for (int k = 0; k < 16; ++k) {
        int c = k * 32 + lane;
        out[base + c] = (v[k] - mean) * rstd * g[c] + b[c];
    }
}

// [b,t,v,D] -> [b,v,t,D]
__global__ __launch_bounds__(256) void permute_btv_bvt(
    const float* __restrict__ src, float* __restrict__ dst)
{
    int idx = blockIdx.x * 256 + threadIdx.x;
    int d = idx & 511;
    int r = idx >> 9;              // bt*7+v
    int v = r % NV;
    int bt = r / NV;
    int b = bt >> 9, t = bt & 511;
    dst[((size_t)(b * NV + v) * NT + t) * DV + d] = src[idx];
}
// [b,v,t,D] -> [b,t, v*D]
__global__ __launch_bounds__(256) void permute_bvt_pool(
    const float* __restrict__ src, float* __restrict__ dst)
{
    int idx = blockIdx.x * 256 + threadIdx.x;
    int d = idx & 511;
    int r = idx >> 9;              // (b*7+v)*512 + t
    int t = r & 511;
    int bv = r >> 9;
    int v = bv % NV, b = bv / NV;
    dst[(size_t)(b * NT + t) * (NV * DV) + v * DV + d] = src[idx];
}

// =====================================================================
extern "C" void kernel_launch(void* const* d_in, const int* in_sizes, int n_in,
                              void* d_out, int out_size, void* d_ws, size_t ws_size,
                              hipStream_t stream) {
    (void)in_sizes; (void)n_in; (void)out_size; (void)ws_size;
    const float* x      = (const float*)d_in[0];
    const float* ve_w   = (const float*)d_in[1];
    const float* ve_b   = (const float*)d_in[2];
    const float* vin_w  = (const float*)d_in[3];
    const float* vin_b  = (const float*)d_in[4];
    const float* vout_w = (const float*)d_in[5];
    const float* vout_b = (const float*)d_in[6];
    const float* tin_w  = (const float*)d_in[7];
    const float* tin_b  = (const float*)d_in[8];
    const float* tout_w = (const float*)d_in[9];
    const float* tout_b = (const float*)d_in[10];
    const float* nvg    = (const float*)d_in[11];
    const float* nvb    = (const float*)d_in[12];
    const float* ntg    = (const float*)d_in[13];
    const float* ntb    = (const float*)d_in[14];
    const float* pw     = (const float*)d_in[15];
    const float* pb     = (const float*)d_in[16];
    const float* ct_in_w  = (const float*)d_in[17];
    const float* ct_in_b  = (const float*)d_in[18];
    const float* ct_out_w = (const float*)d_in[19];
    const float* ct_out_b = (const float*)d_in[20];
    const float* cf_in_w  = (const float*)d_in[21];
    const float* cf_in_b  = (const float*)d_in[22];
    const float* cf_out_w = (const float*)d_in[23];
    const float* cf_out_b = (const float*)d_in[24];
    const float* n1g = (const float*)d_in[25];
    const float* n1b = (const float*)d_in[26];
    const float* n2g = (const float*)d_in[27];
    const float* n2b = (const float*)d_in[28];
    const float* fc1_w = (const float*)d_in[29];
    const float* fc1_b = (const float*)d_in[30];
    const float* fc2_w = (const float*)d_in[31];
    const float* fc2_b = (const float*)d_in[32];
    float* out = (float*)d_out;

    const int BT = NB * NT;          // 4096
    const int MV = BT * NV;          // 28672
    const float scale = 0.08838834764831845f;   // 1/sqrt(128)

    // workspace (floats): f32 buffers + bf16 staging; total ~117.3M floats (~469 MB)
    float* W0 = (float*)d_ws;                  // MV*512
    float* W1 = W0 + (size_t)MV * 512;         // MV*1536
    float* W2 = W1 + (size_t)MV * 1536;        // MV*512
    float* W3 = W2 + (size_t)MV * 512;         // MV*512
    float* W4 = W3 + (size_t)MV * 512;         // MV*512
    float* W5 = W4 + (size_t)MV * 512;         // 3 * BT*512 (fused branch outputs)
    bf16_t* AB = (bf16_t*)(W5 + (size_t)3 * BT * DV);   // bf16 A staging: MV*512
    bf16_t* WB = AB + (size_t)MV * 512;                 // bf16 W staging: 512*3584 max

    auto cvt = [&](const float* src, bf16_t* dst, int elems) {
        cvt_f32_bf16<<<elems / 2048, 256, 0, stream>>>(src, dst);
    };

    for (int br = 0; br < 3; ++br) {
        const float* xb   = x    + (size_t)br * MV;
        const float* vewb = ve_w + (size_t)br * NV * DV;
        const float* vebb = ve_b + (size_t)br * NV * DV;
        float* fused = W5 + (size_t)br * BT * DV;

        // embed -> x1 (W0) [MV, 512]
        embed_kernel<<<MV * DV / 256, 256, 0, stream>>>(xb, vewb, vebb, W0);
        // variant QKV: [MV,512] x [1536,512]^T -> W1
        cvt(W0, AB, MV * 512);
        cvt(vin_w + (size_t)br * 1536 * 512, WB, 1536 * 512);
        gemm_nt_bias<false><<<dim3(1536 / 64, MV / 128), 256, 0, stream>>>(
            AB, 512, WB, 512, vin_b + (size_t)br * 1536, W1, 1536, 512);
        // variant attention -> W2
        vattn_kernel<<<dim3(BT, NH), 128, 0, stream>>>(W1, W2, scale);
        // out proj -> W3
        cvt(W2, AB, MV * 512);
        cvt(vout_w + (size_t)br * 512 * 512, WB, 512 * 512);
        gemm_nt_bias<false><<<dim3(512 / 64, MV / 128), 256, 0, stream>>>(
            AB, 512, WB, 512, vout_b + (size_t)br * 512, W3, 512, 512);
        // LN(x1 + proj) -> W4 (va)
        add_ln_kernel<<<MV / 8, 256, 0, stream>>>(
            W0, W3, nvg + (size_t)br * 512, nvb + (size_t)br * 512, W4);
        // permute [b,t,v,D] -> [b,v,t,D] -> W0 (x2)
        permute_btv_bvt<<<MV * DV / 256, 256, 0, stream>>>(W4, W0);
        // temporal QKV -> W1
        cvt(W0, AB, MV * 512);
        cvt(tin_w + (size_t)br * 1536 * 512, WB, 1536 * 512);
        gemm_nt_bias<false><<<dim3(1536 / 64, MV / 128), 256, 0, stream>>>(
            AB, 512, WB, 512, tin_b + (size_t)br * 1536, W1, 1536, 512);
        // temporal flash attention: 56 seqs of 512 -> W2
        flash_kernel<<<dim3(NT / 8, NH, NB * NV), 256, 0, stream>>>(
            W1,        1536, (long long)NT * 1536,
            W1 + 512,  1536, (long long)NT * 1536,
            W1 + 1024, 1536, (long long)NT * 1536,
            W2, 512, (long long)NT * 512, NT, scale);
        // out proj -> W3
        cvt(W2, AB, MV * 512);
        cvt(tout_w + (size_t)br * 512 * 512, WB, 512 * 512);
        gemm_nt_bias<false><<<dim3(512 / 64, MV / 128), 256, 0, stream>>>(
            AB, 512, WB, 512, tout_b + (size_t)br * 512, W3, 512, 512);
        // LN(x2 + proj) -> W4 (ta, [b,v,t,D])
        add_ln_kernel<<<MV / 8, 256, 0, stream>>>(
            W0, W3, ntg + (size_t)br * 512, ntb + (size_t)br * 512, W4);
        // permute -> pooled input [BT, 3584] -> W0
        permute_bvt_pool<<<MV * DV / 256, 256, 0, stream>>>(W4, W0);
        // pool GEMM: [BT,3584] x [512,3584]^T -> fused[br]
        cvt(W0, AB, BT * NV * DV);
        cvt(pw + (size_t)br * 512 * NV * DV, WB, 512 * NV * DV);
        gemm_nt_bias<false><<<dim3(512 / 64, BT / 128), 256, 0, stream>>>(
            AB, NV * DV, WB, NV * DV, pb + (size_t)br * 512, fused, 512, NV * DV);
    }

    float* t1 = W5;
    float* t2 = W5 + (size_t)BT * DV;
    float* r3 = W5 + (size_t)2 * BT * DV;

    // ---- cross attention 1: q from t1, kv from t2 ----
    cvt(ct_in_w, WB, 1536 * 512);
    cvt(t1, AB, BT * DV);
    gemm_nt_bias<false><<<dim3(512 / 64, BT / 128), 256, 0, stream>>>(
        AB, 512, WB, 512, ct_in_b, W0, 512, 512);
    cvt(t2, AB, BT * DV);
    gemm_nt_bias<false><<<dim3(1024 / 64, BT / 128), 256, 0, stream>>>(
        AB, 512, WB + (size_t)512 * 512, 512, ct_in_b + 512, W1, 1024, 512);
    flash_kernel<<<dim3(NT / 8, NH, NB), 256, 0, stream>>>(
        W0,       512,  (long long)NT * 512,
        W1,       1024, (long long)NT * 1024,
        W1 + 512, 1024, (long long)NT * 1024,
        W2, 512, (long long)NT * 512, NT, scale);
    cvt(W2, AB, BT * DV);
    cvt(ct_out_w, WB, 512 * 512);
    gemm_nt_bias<false><<<dim3(512 / 64, BT / 128), 256, 0, stream>>>(
        AB, 512, WB, 512, ct_out_b, W3, 512, 512);
    add_ln_kernel<<<BT / 8, 256, 0, stream>>>(t1, W3, n1g, n1b, W4);   // W4 = t_f

    // ---- cross attention 2: q from t_f, kv from r3 ----
    cvt(cf_in_w, WB, 1536 * 512);
    cvt(W4, AB, BT * DV);
    gemm_nt_bias<false><<<dim3(512 / 64, BT / 128), 256, 0, stream>>>(
        AB, 512, WB, 512, cf_in_b, W0, 512, 512);
    cvt(r3, AB, BT * DV);
    gemm_nt_bias<false><<<dim3(1024 / 64, BT / 128), 256, 0, stream>>>(
        AB, 512, WB + (size_t)512 * 512, 512, cf_in_b + 512, W1, 1024, 512);
    flash_kernel<<<dim3(NT / 8, NH, NB), 256, 0, stream>>>(
        W0,       512,  (long long)NT * 512,
        W1,       1024, (long long)NT * 1024,
        W1 + 512, 1024, (long long)NT * 1024,
        W2, 512, (long long)NT * 512, NT, scale);
    cvt(W2, AB, BT * DV);
    cvt(cf_out_w, WB, 512 * 512);
    gemm_nt_bias<false><<<dim3(512 / 64, BT / 128), 256, 0, stream>>>(
        AB, 512, WB, 512, cf_out_b, W3, 512, 512);
    add_ln_kernel<<<BT / 8, 256, 0, stream>>>(W4, W3, n2g, n2b, W0);   // W0 = out_ln

    // ---- MLP head ----
    cvt(W0, AB, BT * DV);
    cvt(fc1_w, WB, 512 * 512);
    gemm_nt_bias<true><<<dim3(512 / 64, BT / 128), 256, 0, stream>>>(
        AB, 512, WB, 512, fc1_b, W1, 512, 512);                        // relu
    cvt(W1, AB, BT * DV);
    cvt(fc2_w, WB, 512 * 512);
    gemm_nt_bias<false><<<dim3(512 / 64, BT / 128), 256, 0, stream>>>(
        AB, 512, WB, 512, fc2_b, out, 512, 512);
}
